// Chowder_558345749022
// MI455X (gfx1250) — matrix operand order repaired and checked by reading, not compiled, not run on hardware
//
#include <hip/hip_runtime.h>
#include <math.h>

// ---------------------------------------------------------------------------
// Chowder MIL model for MI455X (gfx1250, wave32, WMMA).
//   1) scores[b,n] = sw2 . sigmoid(sw1 @ x[b,n] + sb1) + sb2   (bf16x3 WMMA)
//   2) per-bag exact top-100 / bottom-100 via LDS bitonic sort
//   3) tiny 3-layer MLP head in f32 VALU
// ---------------------------------------------------------------------------

typedef __attribute__((ext_vector_type(16))) __bf16 v16bf;
typedef __attribute__((ext_vector_type(8)))  __bf16 v8bf;
typedef __attribute__((ext_vector_type(8)))  float  v8f;

#define BAGS      32
#define N_INST    10000
#define D_IN      768
#define D_PAD     771      // 768 features + 3 meta columns
#define H1        128
#define ROWS      (BAGS * N_INST)   // 320000
#define N_TOPK    100
#define N_EXT     200

__device__ __forceinline__ float sigmoidf(float x) {
    return 1.0f / (1.0f + expf(-x));
}

// ---------------------------------------------------------------------------
// Kernel 0: split sw1 (128x768 f32) into bf16 hi/lo parts (bf16x3 scheme).
// ---------------------------------------------------------------------------
__global__ void k_split_w(const float* __restrict__ w, __bf16* __restrict__ hi,
                          __bf16* __restrict__ lo, int n) {
    int i = blockIdx.x * blockDim.x + threadIdx.x;
    if (i < n) {
        float v = w[i];
        __bf16 h = (__bf16)v;
        hi[i] = h;
        lo[i] = (__bf16)(v - (float)h);
    }
}

// ---------------------------------------------------------------------------
// Kernel 1: big GEMM + sigmoid + score dot, one 16-row tile per wave.
// A (16x32 bf16): lane l(0-15) holds row M=l, K {0..7,16..23};
//                 lane l+16 holds row M=l, K {8..15,24..31}.
// B (32x16 bf16): lane n(0-15) holds col N=n, K {0..7,16..23}; lane n+16 K {8..15,24..31}.
// C (16x16 f32) : lane l holds N=l%16, rows M = 8*(l/16) + vgpr(0..7).
// ---------------------------------------------------------------------------
__global__ void __launch_bounds__(256)
k_scores(const float* __restrict__ feat,
         const __bf16* __restrict__ w1h, const __bf16* __restrict__ w1l,
         const float* __restrict__ sb1,  const float* __restrict__ sw2,
         const float* __restrict__ sb2,  float* __restrict__ scores)
{
    const int lane = threadIdx.x & 31;
    const int wave = threadIdx.x >> 5;
    const int tile = blockIdx.x * 8 + wave;        // 20000 tiles of 16 rows
    const int row0 = tile * 16;
    const int hf   = lane >> 4;                    // which half of the wave
    const int lr   = lane & 15;

    // this lane's A row (skip 3 meta columns)
    const float* arow = feat + (size_t)(row0 + lr) * D_PAD + 3;

    v8f acc[8] = {};                               // 16 rows x 128 hidden

    for (int k0 = 0; k0 < D_IN; k0 += 32) {
        // ---- load + split A tile (16 f32 per lane, two 32B runs) ----
        float a[16];
        const float* p0 = arow + k0 + hf * 8;
        #pragma unroll
        for (int i = 0; i < 8; ++i) a[i] = p0[i];
        #pragma unroll
        for (int i = 0; i < 8; ++i) a[8 + i] = p0[16 + i];

        v16bf Ah, Al;
        #pragma unroll
        for (int i = 0; i < 16; ++i) {
            __bf16 h = (__bf16)a[i];
            Ah[i] = h;
            Al[i] = (__bf16)(a[i] - (float)h);
        }

        // ---- 8 hidden tiles of 16, three WMMAs each (bf16x3) ----
        #pragma unroll
        for (int nt = 0; nt < 8; ++nt) {
            const size_t boff = (size_t)(nt * 16 + lr) * D_IN + k0 + hf * 8;
            v16bf Bh, Bl;
            {
                v8bf b0 = *(const v8bf*)(w1h + boff);
                v8bf b1 = *(const v8bf*)(w1h + boff + 16);
                #pragma unroll
                for (int i = 0; i < 8; ++i) { Bh[i] = b0[i]; Bh[i + 8] = b1[i]; }
            }
            {
                v8bf b0 = *(const v8bf*)(w1l + boff);
                v8bf b1 = *(const v8bf*)(w1l + boff + 16);
                #pragma unroll
                for (int i = 0; i < 8; ++i) { Bl[i] = b0[i]; Bl[i + 8] = b1[i]; }
            }
            acc[nt] = __builtin_amdgcn_wmma_f32_16x16x32_bf16(
                false, Ah, false, Bh, (short)0, acc[nt], false, false);
            acc[nt] = __builtin_amdgcn_wmma_f32_16x16x32_bf16(
                false, Ah, false, Bl, (short)0, acc[nt], false, false);
            acc[nt] = __builtin_amdgcn_wmma_f32_16x16x32_bf16(
                false, Al, false, Bh, (short)0, acc[nt], false, false);
        }
    }

    // ---- epilogue: bias + sigmoid + dot with sw2, fused per lane ----
    float part[8] = {};
    #pragma unroll
    for (int nt = 0; nt < 8; ++nt) {
        const int j = nt * 16 + lr;                // hidden unit of this lane
        const float b1 = sb1[j];
        const float w2 = sw2[j];
        #pragma unroll
        for (int r = 0; r < 8; ++r)
            part[r] += sigmoidf(acc[nt][r] + b1) * w2;
    }

    // reduce over the 16 lanes of each half-wave (sums the 128 hidden units)
    #pragma unroll
    for (int m = 1; m < 16; m <<= 1) {
        #pragma unroll
        for (int r = 0; r < 8; ++r)
            part[r] += __shfl_xor(part[r], m, 32);
    }

    if (lr == 0) {
        const float b2 = sb2[0];
        #pragma unroll
        for (int r = 0; r < 8; ++r)
            scores[row0 + hf * 8 + r] = part[r] + b2;
    }
}

// ---------------------------------------------------------------------------
// Kernel 2: per-bag exact selection. Bitonic-sort 16384 padded values
// descending in 64KB LDS; emit top-100 (desc) then bottom-100 (asc).
// ---------------------------------------------------------------------------
#define TK_N 16384
#define TK_T 512

__global__ void __launch_bounds__(TK_T)
k_topk(const float* __restrict__ scores, float* __restrict__ extreme)
{
    __shared__ float s[TK_N];
    const int b = blockIdx.x;
    const float* sc = scores + (size_t)b * N_INST;

    for (int i = threadIdx.x; i < TK_N; i += TK_T)
        s[i] = (i < N_INST) ? sc[i] : -3.402823466e38f;
    __syncthreads();

    for (int k = 2; k <= TK_N; k <<= 1) {
        for (int j = k >> 1; j > 0; j >>= 1) {
            for (int i = threadIdx.x; i < TK_N; i += TK_T) {
                const int ixj = i ^ j;
                if (ixj > i) {
                    const float a = s[i], c = s[ixj];
                    const bool desc = ((i & k) == 0);
                    if (desc ? (a < c) : (a > c)) { s[i] = c; s[ixj] = a; }
                }
            }
            __syncthreads();
        }
    }

    float* e = extreme + b * N_EXT;
    if (threadIdx.x < N_TOPK)
        e[threadIdx.x] = s[threadIdx.x];                       // top, descending
    else if (threadIdx.x < N_EXT)
        e[threadIdx.x] = s[(N_INST - 1) - (threadIdx.x - 100)]; // bottom, ascending
}

// ---------------------------------------------------------------------------
// Kernel 3: 200 -> 128 -> 64 -> 1 MLP head, one block per bag (f32 exact).
// ---------------------------------------------------------------------------
__global__ void __launch_bounds__(128)
k_mlp(const float* __restrict__ extreme,
      const float* __restrict__ mw1, const float* __restrict__ mb1,
      const float* __restrict__ mw2, const float* __restrict__ mb2,
      const float* __restrict__ mw3, const float* __restrict__ mb3,
      float* __restrict__ out)
{
    __shared__ float e[N_EXT];
    __shared__ float h1[128];
    __shared__ float h2[64];
    const int b = blockIdx.x, t = threadIdx.x;

    for (int i = t; i < N_EXT; i += 128) e[i] = extreme[b * N_EXT + i];
    __syncthreads();

    {
        float acc = mb1[t];
        const float* wr = mw1 + t * N_EXT;
        #pragma unroll 8
        for (int k = 0; k < N_EXT; ++k) acc += wr[k] * e[k];
        h1[t] = sigmoidf(acc);
    }
    __syncthreads();

    if (t < 64) {
        float acc = mb2[t];
        const float* wr = mw2 + t * 128;
        #pragma unroll 8
        for (int k = 0; k < 128; ++k) acc += wr[k] * h1[k];
        h2[t] = sigmoidf(acc);
    }
    __syncthreads();

    if (t == 0) {
        float y = mb3[0];
        #pragma unroll 8
        for (int g = 0; g < 64; ++g) y += mw3[g] * h2[g];
        out[b] = y;
    }
}

// ---------------------------------------------------------------------------
extern "C" void kernel_launch(void* const* d_in, const int* in_sizes, int n_in,
                              void* d_out, int out_size, void* d_ws, size_t ws_size,
                              hipStream_t stream)
{
    (void)in_sizes; (void)n_in; (void)out_size; (void)ws_size;

    const float* features = (const float*)d_in[0];
    const float* sw1      = (const float*)d_in[1];
    const float* sb1      = (const float*)d_in[2];
    const float* sw2      = (const float*)d_in[3];
    const float* sb2      = (const float*)d_in[4];
    const float* mw1      = (const float*)d_in[5];
    const float* mb1      = (const float*)d_in[6];
    const float* mw2      = (const float*)d_in[7];
    const float* mb2      = (const float*)d_in[8];
    const float* mw3      = (const float*)d_in[9];
    const float* mb3      = (const float*)d_in[10];

    // workspace layout (all 256B aligned)
    char* ws = (char*)d_ws;
    float*  scores  = (float*)ws;                                   // 1,280,000 B
    __bf16* w1h     = (__bf16*)(ws + 1280000);                      //   196,608 B
    __bf16* w1l     = (__bf16*)(ws + 1280000 + 196608);             //   196,608 B
    float*  extreme = (float*)(ws + 1280000 + 2 * 196608);          //    25,600 B

    const int NW = H1 * D_IN;                                       // 98304
    k_split_w<<<(NW + 255) / 256, 256, 0, stream>>>(sw1, w1h, w1l, NW);

    k_scores<<<ROWS / 16 / 8, 256, 0, stream>>>(features, w1h, w1l,
                                                sb1, sw2, sb2, scores);

    k_topk<<<BAGS, TK_T, 0, stream>>>(scores, extreme);

    k_mlp<<<BAGS, 128, 0, stream>>>(extreme, mw1, mb1, mw2, mb2, mw3, mb3,
                                    (float*)d_out);
}
